// InstanceEmbedding_71760313582001
// MI455X (gfx1250) — matrix operand order repaired and checked by reading, not compiled, not run on hardware
//
#include <hip/hip_runtime.h>
#include <hip/hip_bf16.h>

// ---------------- problem constants ----------------
constexpr int BW      = 256;     // B*W
constexpr int M_TOT   = 2560;    // metrics
constexpr int D       = 256;     // D_IN == D_MODEL
constexpr int NPODS   = 32;
constexpr long ROWS   = (long)BW * M_TOT;   // 655360

typedef __attribute__((ext_vector_type(16))) __bf16 v16bf;
typedef __attribute__((ext_vector_type(8)))  __bf16 v8bf;
typedef __attribute__((ext_vector_type(8)))  float  v8f;

// ============================================================
// Kernel 0: swizzle W_proj / W1 (fp32 -> bf16) into WMMA
// B-fragment order:  [ntile(16)][kstep(8)][lane(32)][j(16)]
//   lane 0-15 : N = ntile*16 + lane,    K = kstep*32 + (j&7) + 16*(j>>3)       (j=0..15)
//   lane 16-31: N = ntile*16 + lane-16, K = kstep*32 + 8 + (j&7) + 16*(j>>3)
// ============================================================
__global__ __launch_bounds__(256) void k_swizzle(const float* __restrict__ Wp,
                                                 const float* __restrict__ W1,
                                                 __bf16* __restrict__ WpSw,
                                                 __bf16* __restrict__ W1Sw) {
    int t   = blockIdx.x * 256 + threadIdx.x;          // 0 .. 131071
    int mat = t >> 16;
    int e   = t & 65535;
    int j    =  e        & 15;
    int lane = (e >> 4)  & 31;
    int kk   = (e >> 9)  & 7;
    int ntg  =  e >> 12;                               // 0..15
    int n = ntg * 16 + (lane & 15);
    int k = kk * 32 + ((lane >> 4) * 8) + (j & 7) + ((j >> 3) * 16);
    const float* src = mat ? W1   : Wp;
    __bf16*      dst = mat ? W1Sw : WpSw;
    dst[e] = (__bf16)src[k * D + n];
}

// ============================================================
// Kernel B: segment bounds[p] = lower_bound(seg_id, p), p=0..32
// ============================================================
__global__ void k_bounds(const int* __restrict__ seg, int* __restrict__ bounds) {
    int p = threadIdx.x;
    if (p > NPODS) return;
    int lo = 0, hi = M_TOT;
    while (lo < hi) {
        int mid = (lo + hi) >> 1;
        if (seg[mid] < p) lo = mid + 1; else hi = mid;
    }
    bounds[p] = lo;
}

// ============================================================
// Kernel 1: fused  proj = x@Wp + bp  (bf16 WMMA, f32 acc)
//                  h    = gelu(proj@W1 + b1)
//                  logit= h@W2 + b2
// 64 rows / block, 256 thr = 8 waves; wave = 16 rows x 128 cols.
// ============================================================
__global__ __launch_bounds__(256) void k_fused(const float*  __restrict__ x,
                                               const __bf16* __restrict__ WpSw,
                                               const __bf16* __restrict__ W1Sw,
                                               const float*  __restrict__ bproj,
                                               const float*  __restrict__ b1,
                                               const float*  __restrict__ W2,
                                               const float*  __restrict__ b2,
                                               __bf16* __restrict__ projOut,
                                               float*  __restrict__ logitsOut) {
    __shared__ __bf16 projLDS[64 * D];     // 32 KB, A-source for GEMM2 + store staging
    __shared__ float  logitsBuf[64];

    const int  tid      = threadIdx.x;
    const int  lane     = tid & 31;
    const int  wave     = tid >> 5;
    const int  rowGroup = wave & 3;        // 4 row groups x 16 rows
    const int  colGroup = wave >> 2;       // 2 col groups x 128 cols
    const int  l15      = lane & 15;
    const int  lhi      = lane >> 4;       // 0 | 1
    const long blkRow0  = (long)blockIdx.x * 64;
    const long rowBase  = blkRow0 + rowGroup * 16;

    if (tid < 64) logitsBuf[tid] = b2[0];

    const v8f vzero = {0.f, 0.f, 0.f, 0.f, 0.f, 0.f, 0.f, 0.f};

    // ---------------- GEMM1: proj = x @ Wp ----------------
    v8f acc[8];
#pragma unroll
    for (int nt = 0; nt < 8; ++nt) acc[nt] = vzero;

    const float* xrow = x + (rowBase + l15) * D;
#pragma unroll
    for (int kk = 0; kk < 8; ++kk) {
        const int kb = kk * 32 + lhi * 8;   // per-lane K base (A layout)
        const float4* q0 = (const float4*)(xrow + kb);
        const float4* q1 = (const float4*)(xrow + kb + 16);
        float4 f0 = q0[0], f1 = q0[1], f2 = q1[0], f3 = q1[1];
        v16bf a;
        a[0]  = (__bf16)f0.x; a[1]  = (__bf16)f0.y; a[2]  = (__bf16)f0.z; a[3]  = (__bf16)f0.w;
        a[4]  = (__bf16)f1.x; a[5]  = (__bf16)f1.y; a[6]  = (__bf16)f1.z; a[7]  = (__bf16)f1.w;
        a[8]  = (__bf16)f2.x; a[9]  = (__bf16)f2.y; a[10] = (__bf16)f2.z; a[11] = (__bf16)f2.w;
        a[12] = (__bf16)f3.x; a[13] = (__bf16)f3.y; a[14] = (__bf16)f3.z; a[15] = (__bf16)f3.w;
#pragma unroll
        for (int nt = 0; nt < 8; ++nt) {
            const int ntg = colGroup * 8 + nt;
            v16bf bf = *(const v16bf*)(WpSw + ((((ntg * 8 + kk) * 32) + lane) << 4));
            acc[nt] = __builtin_amdgcn_wmma_f32_16x16x32_bf16(
                false, a, false, bf, (short)0, acc[nt], false, false);
        }
    }

    // bias, write proj tile to LDS (A-layout source for GEMM2 + store staging)
#pragma unroll
    for (int nt = 0; nt < 8; ++nt) {
        const int ntg  = colGroup * 8 + nt;
        const int n    = ntg * 16 + l15;
        const float bs = bproj[n];
#pragma unroll
        for (int i = 0; i < 8; ++i) {
            const float v  = acc[nt][i] + bs;
            const int rloc = rowGroup * 16 + i + lhi * 8;   // C layout: row i / i+8
            projLDS[rloc * D + n] = (__bf16)v;
        }
    }
    __syncthreads();

    // coalesced tile store: 64-row tile is contiguous in projOut -> flat 32KB copy
    // thread t moves 8 x 16B chunks; consecutive lanes hit contiguous 512B runs
    {
        __bf16* gbase = projOut + blkRow0 * D;
#pragma unroll
        for (int u = 0; u < 8; ++u) {
            const int off = (u * 256 + tid) * 8;   // element offset (16B aligned)
            *(v8bf*)(gbase + off) = *(const v8bf*)(projLDS + off);
        }
    }

    // ---------------- GEMM2: h = gelu(proj @ W1 + b1) ----------------
    v8f acc2[8];
#pragma unroll
    for (int nt = 0; nt < 8; ++nt) acc2[nt] = vzero;

    const __bf16* prow = projLDS + (rowGroup * 16 + l15) * D;
#pragma unroll
    for (int kk = 0; kk < 8; ++kk) {
        const int kb = kk * 32 + lhi * 8;
        v8bf lo = *(const v8bf*)(prow + kb);
        v8bf hi = *(const v8bf*)(prow + kb + 16);
        v16bf a;
#pragma unroll
        for (int j = 0; j < 8; ++j) { a[j] = lo[j]; a[8 + j] = hi[j]; }
#pragma unroll
        for (int nt = 0; nt < 8; ++nt) {
            const int ntg = colGroup * 8 + nt;
            v16bf bf = *(const v16bf*)(W1Sw + ((((ntg * 8 + kk) * 32) + lane) << 4));
            acc2[nt] = __builtin_amdgcn_wmma_f32_16x16x32_bf16(
                false, a, false, bf, (short)0, acc2[nt], false, false);
        }
    }

    // GELU (exact, erf) + logits partial = sum_n h[m,n] * W2[n]
    float part[8];
#pragma unroll
    for (int i = 0; i < 8; ++i) part[i] = 0.f;
#pragma unroll
    for (int nt = 0; nt < 8; ++nt) {
        const int n    = (colGroup * 8 + nt) * 16 + l15;
        const float bb = b1[n];
        const float wn = W2[n];
#pragma unroll
        for (int i = 0; i < 8; ++i) {
            const float v = acc2[nt][i] + bb;
            const float g = 0.5f * v * (1.0f + erff(v * 0.70710678118654752f));
            part[i] += g * wn;
        }
    }
    // reduce over the 16 lanes of each half (rows i vs i+8 live in halves)
#pragma unroll
    for (int mk = 1; mk <= 8; mk <<= 1)
#pragma unroll
        for (int i = 0; i < 8; ++i) part[i] += __shfl_xor(part[i], mk, 32);
    if (l15 == 0) {
#pragma unroll
        for (int i = 0; i < 8; ++i)
            atomicAdd(&logitsBuf[rowGroup * 16 + lhi * 8 + i], part[i]);
    }
    __syncthreads();
    if (tid < 64) logitsOut[blkRow0 + tid] = logitsBuf[tid];
}

// ============================================================
// Kernel 2: per-(bw) segmented softmax over M.
// ============================================================
__global__ __launch_bounds__(256) void k_softmax(const float* __restrict__ logits,
                                                 const int*   __restrict__ seg,
                                                 float* __restrict__ attn) {
    __shared__ unsigned pmax[NPODS];
    __shared__ float    psum[NPODS];
    const int tid = threadIdx.x;
    const int bw  = blockIdx.x;
    if (tid < NPODS) { pmax[tid] = 0u; psum[tid] = 0.f; }
    __syncthreads();

    const float* lrow = logits + (long)bw * M_TOT;
    float*       arow = attn   + (long)bw * M_TOT;

    for (int m = tid; m < M_TOT; m += 256) {
        const unsigned b   = __float_as_uint(lrow[m]);
        const unsigned key = (b & 0x80000000u) ? ~b : (b | 0x80000000u);
        atomicMax(&pmax[seg[m]], key);
    }
    __syncthreads();
    for (int m = tid; m < M_TOT; m += 256) {
        const unsigned key = pmax[seg[m]];
        const unsigned b   = (key & 0x80000000u) ? (key & 0x7FFFFFFFu) : ~key;
        const float e = expf(lrow[m] - __uint_as_float(b));
        arow[m] = e;
        atomicAdd(&psum[seg[m]], e);
    }
    __syncthreads();
    for (int m = tid; m < M_TOT; m += 256) arow[m] /= psum[seg[m]];
}

// ============================================================
// Kernel 3: agg[bw,p,d] = sum_{m in seg p} attn[bw,m]*proj[bw,m,d]
// one block per (bw,pod); 128 thr, 2 d's / thread (bf16x2 loads)
// ============================================================
__global__ __launch_bounds__(128) void k_agg(const __bf16* __restrict__ proj,
                                             const float*  __restrict__ attn,
                                             const int*    __restrict__ bounds,
                                             float* __restrict__ out) {
    const int bw = blockIdx.x >> 5;
    const int p  = blockIdx.x & 31;
    const int d0 = threadIdx.x * 2;
    const int s  = bounds[p];
    const int e  = bounds[p + 1];
    const float* arow = attn + (long)bw * M_TOT;
    const __bf16* prow0 = proj + (long)bw * M_TOT * D + d0;
    float a0 = 0.f, a1 = 0.f;
    for (int m = s; m < e; ++m) {
        // hide HBM latency of the serial segment walk (-> global_prefetch_b8)
        __builtin_prefetch(prow0 + (long)(m + 16) * D, 0, 1);
        const float w = arow[m];
        const unsigned pv = *(const unsigned*)(prow0 + (long)m * D);
        a0 += w * __uint_as_float(pv << 16);
        a1 += w * __uint_as_float(pv & 0xFFFF0000u);
    }
    const long o = ((long)bw * NPODS + p) * D + d0;
    out[o]     = a0;
    out[o + 1] = a1;
}

// ============================================================
extern "C" void kernel_launch(void* const* d_in, const int* in_sizes, int n_in,
                              void* d_out, int out_size, void* d_ws, size_t ws_size,
                              hipStream_t stream) {
    const float* x     = (const float*)d_in[0];
    const int*   seg   = (const int*)  d_in[1];
    const float* Wp    = (const float*)d_in[2];
    const float* bp    = (const float*)d_in[3];
    const float* W1    = (const float*)d_in[4];
    const float* b1    = (const float*)d_in[5];
    const float* W2    = (const float*)d_in[6];
    const float* b2    = (const float*)d_in[7];
    float*       out   = (float*)d_out;

    // workspace layout
    char* w = (char*)d_ws;
    __bf16* proj   = (__bf16*)w;                              // 335,544,320 B
    size_t  off    = (size_t)ROWS * D * sizeof(__bf16);
    float*  logits = (float*)(w + off);   off += (size_t)ROWS * sizeof(float);
    float*  attn   = (float*)(w + off);   off += (size_t)ROWS * sizeof(float);
    __bf16* WpSw   = (__bf16*)(w + off);  off += (size_t)D * D * sizeof(__bf16);
    __bf16* W1Sw   = (__bf16*)(w + off);  off += (size_t)D * D * sizeof(__bf16);
    int*    bounds = (int*)(w + off);

    k_swizzle<<<dim3(512), dim3(256), 0, stream>>>(Wp, W1, WpSw, W1Sw);
    k_bounds <<<dim3(1),   dim3(64),  0, stream>>>(seg, bounds);
    k_fused  <<<dim3((unsigned)(ROWS / 64)), dim3(256), 0, stream>>>(
        x, WpSw, W1Sw, bp, b1, W2, b2, proj, logits);
    k_softmax<<<dim3(BW),  dim3(256), 0, stream>>>(logits, seg, attn);
    k_agg    <<<dim3(BW * NPODS), dim3(128), 0, stream>>>(proj, attn, bounds, out);
}
